// PointnetSAModuleMSG_58669253263780
// MI455X (gfx1250) — matrix operand order repaired and checked by reading, not compiled, not run on hardware
//
#include <hip/hip_runtime.h>
#include <math.h>

// ---------------------------------------------------------------------------
// Problem constants (match reference)
// ---------------------------------------------------------------------------
constexpr int Bn  = 4;
constexpr int Nn  = 16384;
constexpr int CIN = 64;
constexpr int NP  = 2048;

// ---------------------------------------------------------------------------
// WMMA types (gfx1250, wave32): v_wmma_f32_16x16x32_bf16
// ---------------------------------------------------------------------------
typedef __attribute__((ext_vector_type(16))) __bf16       v16bf;
typedef __attribute__((ext_vector_type(4)))  __bf16       v4bf;
typedef __attribute__((ext_vector_type(8)))  float        v8f;
typedef __attribute__((ext_vector_type(4)))  unsigned int v4u;

union Frag { v16bf v; v4u q[2]; };

// ---------------------------------------------------------------------------
// Kernel 1: furthest point sampling. One block per batch, 1024 threads,
// per-thread distance array in registers, two-stage argmax (wave32 shuffle
// then 32-wide LDS reduce). Writes fps indices + new_xyz (to d_out).
// ---------------------------------------------------------------------------
__global__ __launch_bounds__(1024) void fps_kernel(const float* __restrict__ xyz,
                                                   int* __restrict__ fpsidx,
                                                   float* __restrict__ newxyz) {
    const int b = blockIdx.x, t = threadIdx.x;
    constexpr int PT = Nn / 1024;    // 16 points per thread
    float px[PT], py[PT], pz[PT], dd[PT];
#pragma unroll
    for (int i = 0; i < PT; ++i) {
        int g = t + i * 1024;
        px[i] = xyz[(b * Nn + g) * 3 + 0];
        py[i] = xyz[(b * Nn + g) * 3 + 1];
        pz[i] = xyz[(b * Nn + g) * 3 + 2];
        dd[i] = 1e10f;
    }
    __shared__ float sv[32];
    __shared__ int   si[32];
    __shared__ int   sfar;
    const int lane = t & 31, wv = t >> 5;
    int far = 0;
    for (int j = 0; j < NP; ++j) {
        float cx = xyz[(b * Nn + far) * 3 + 0];
        float cy = xyz[(b * Nn + far) * 3 + 1];
        float cz = xyz[(b * Nn + far) * 3 + 2];
        if (t == 0) {
            fpsidx[b * NP + j] = far;
            newxyz[(b * NP + j) * 3 + 0] = cx;
            newxyz[(b * NP + j) * 3 + 1] = cy;
            newxyz[(b * NP + j) * 3 + 2] = cz;
        }
        float best = -1.0f; int bi = 0x7fffffff;
#pragma unroll
        for (int i = 0; i < PT; ++i) {
            float dx = px[i] - cx, dy = py[i] - cy, dz = pz[i] - cz;
            float d = dx * dx + dy * dy + dz * dz;
            dd[i] = fminf(dd[i], d);
            if (dd[i] > best) { best = dd[i]; bi = t + i * 1024; }
        }
#pragma unroll
        for (int off = 16; off; off >>= 1) {
            float ov = __shfl_down(best, off);
            int   oi = __shfl_down(bi,   off);
            if (ov > best || (ov == best && oi < bi)) { best = ov; bi = oi; }
        }
        if (lane == 0) { sv[wv] = best; si[wv] = bi; }
        __syncthreads();
        if (t < 32) {
            best = sv[t]; bi = si[t];
#pragma unroll
            for (int off = 16; off; off >>= 1) {
                float ov = __shfl_down(best, off);
                int   oi = __shfl_down(bi,   off);
                if (ov > best || (ov == best && oi < bi)) { best = ov; bi = oi; }
            }
            if (t == 0) sfar = bi;
        }
        __syncthreads();
        far = sfar;
    }
}

// ---------------------------------------------------------------------------
// Kernel 2: ball query. One wave per query point; 32-wide ordered scan with
// ballot + prefix popcount reproduces "smallest nsample indices in radius".
// ---------------------------------------------------------------------------
template <int SAMP>
__global__ __launch_bounds__(256) void ball_kernel(const float* __restrict__ xyz,
                                                   const float* __restrict__ newxyz,
                                                   int* __restrict__ idxout,
                                                   float r2) {
    const int lane = threadIdx.x & 31;
    const int wid  = threadIdx.x >> 5;
    const int q    = blockIdx.x * 8 + wid;      // [0, B*NP)
    const int b    = q >> 11;
    float qx = newxyz[q * 3 + 0], qy = newxyz[q * 3 + 1], qz = newxyz[q * 3 + 2];
    int* myout = idxout + (size_t)q * SAMP;
    int cnt = 0;
    for (int base = 0; base < Nn && cnt < SAMP; base += 32) {
        int i = base + lane;
        float dx = xyz[(b * Nn + i) * 3 + 0] - qx;
        float dy = xyz[(b * Nn + i) * 3 + 1] - qy;
        float dz = xyz[(b * Nn + i) * 3 + 2] - qz;
        bool hit = (dx * dx + dy * dy + dz * dz) < r2;
        unsigned m = (unsigned)__ballot(hit);
        int pos = cnt + __popc(m & ((1u << lane) - 1u));
        if (hit && pos < SAMP) myout[pos] = i;
        cnt += __popc(m);
    }
    if (cnt < SAMP) {
        int first = (cnt > 0) ? myout[0] : 0;   // same-wave store->load is ordered
        for (int k = cnt + lane; k < SAMP; k += 32) myout[k] = first;
    }
}

// ---------------------------------------------------------------------------
// Kernel 3: grouped MLP via WMMA bf16 (fp32 accumulate), recompute-based BN.
//   PASS 0: Y1 = X @ W1^T              -> per-channel sum/sumsq partials
//   PASS 1: H1 = bnrelu(Y1); Y2 = H1W2 -> per-channel sum/sumsq partials
//   PASS 2: full chain + bnrelu2 + max over S -> output slice
// 2 waves / block, each wave processes QPW query points sequentially
// (amortizes LDS weight staging). X layout: [features(64) | relxyz(3) | pad];
// W1 is permuted identically at staging time, so Y is unchanged.
// Fragment layouts per CDNA5 ISA §7.12.2 (16-bit A 16x32; B K-major rows).
// ---------------------------------------------------------------------------
template <int SAMP, int C1, int KP1, int C2, int PASS, int COFF, int QPW>
__global__ __launch_bounds__(64) void mlp_kernel(
    const float* __restrict__ xyz, const float* __restrict__ feat,
    const float* __restrict__ newxyz, const int* __restrict__ bidx,
    const float* __restrict__ w1, const float* __restrict__ w2,
    const float* __restrict__ ss1, const float* __restrict__ ss2,
    float* __restrict__ statout, float* __restrict__ outfeat) {

    constexpr int KP2 = C1;   // layer-2 K (64 or 96, both multiples of 32)

    __shared__ __attribute__((aligned(16))) __bf16 W1s[C1][KP1];
    __shared__ __attribute__((aligned(16))) __bf16 W2s[C2][KP2];
    __shared__ __attribute__((aligned(16))) __bf16 As[2][16][KP1];
    __shared__ __attribute__((aligned(16))) __bf16 Hs[2][16][KP2];
    __shared__ float stat[2][256];     // [wave][sum(0..127) | sumsq(128..255)]

    const int tid = threadIdx.x;
    // Stage layer-1 weights with the permuted K layout:
    //   staged k in [0,64)  <- src (k+3)   (features)
    //   staged k in [64,67) <- src (k-64)  (xyz)
    //   staged k >= 67      <- 0           (pad)
    for (int i = tid; i < C1 * KP1; i += 64) {
        int o = i / KP1, k = i - o * KP1;
        float v = 0.f;
        if (k < 64)      v = w1[o * 67 + (k + 3)];
        else if (k < 67) v = w1[o * 67 + (k - 64)];
        W1s[o][k] = (__bf16)v;
    }
    if constexpr (PASS >= 1) {
        for (int i = tid; i < C2 * KP2; i += 64) {
            int o = i / KP2, k = i - o * KP2;
            W2s[o][k] = (__bf16)w2[o * C1 + k];
        }
    }
    if constexpr (PASS <= 1) {
        for (int i = tid; i < 2 * 256; i += 64) ((float*)stat)[i] = 0.f;
    }
    __syncthreads();

    const int w = tid >> 5, lane = tid & 31;
    const int g16 = lane >> 4;    // lane group (0: lanes 0-15, 1: lanes 16-31)
    const int l16 = lane & 15;

    // Zero the constant K-pad region of this wave's A-tile once.
    for (int r = 0; r < 16; ++r)
        for (int c = 67 + lane; c < KP1; c += 32)
            As[w][r][c] = (__bf16)0.f;

    for (int qi = 0; qi < QPW; ++qi) {
        const int q = blockIdx.x * (2 * QPW) + w * QPW + qi;
        const int b = q >> 11, p = q & (NP - 1);
        const float qx = newxyz[q * 3 + 0], qy = newxyz[q * 3 + 1], qz = newxyz[q * 3 + 2];

        float cmax[C2 / 16];
#pragma unroll
        for (int i = 0; i < C2 / 16; ++i) cmax[i] = -1e30f;

        for (int t = 0; t < SAMP / 16; ++t) {
            // ---- gather 16 rows, two rows per step (lane group = row parity).
            // Each lane: one float4 feature load -> 4 bf16 -> one b64 LDS store.
#pragma unroll
            for (int pr = 0; pr < 8; ++pr) {
                const int row = pr * 2 + g16;
                const int j = bidx[(size_t)q * SAMP + t * 16 + row];
                const float4 f4 =
                    *(const float4*)(feat + (size_t)(b * Nn + j) * CIN + l16 * 4);
                v4bf h;
                h[0] = (__bf16)f4.x; h[1] = (__bf16)f4.y;
                h[2] = (__bf16)f4.z; h[3] = (__bf16)f4.w;
                *(v4bf*)&As[w][row][l16 * 4] = h;
                if (l16 < 3) {
                    float c = xyz[(size_t)(b * Nn + j) * 3 + l16] -
                              (l16 == 0 ? qx : (l16 == 1 ? qy : qz));
                    As[w][row][64 + l16] = (__bf16)c;
                }
            }
            // ---- layer 1: Y1[16 x C1] ----
#pragma unroll
            for (int n1 = 0; n1 < C1 / 16; ++n1) {
                v8f acc = {0.f, 0.f, 0.f, 0.f, 0.f, 0.f, 0.f, 0.f};
#pragma unroll
                for (int ks = 0; ks < KP1 / 32; ++ks) {
                    Frag fa, fb;
                    const __bf16* pa = &As[w][l16][ks * 32 + g16 * 8];
                    fa.q[0] = *(const v4u*)pa;
                    fa.q[1] = *(const v4u*)(pa + 16);
                    const __bf16* pb = &W1s[n1 * 16 + l16][ks * 32 + g16 * 8];
                    fb.q[0] = *(const v4u*)pb;
                    fb.q[1] = *(const v4u*)(pb + 16);
                    acc = __builtin_amdgcn_wmma_f32_16x16x32_bf16(
                        false, fa.v, false, fb.v, (short)0, acc, false, false);
                }
                const int ch = n1 * 16 + l16;
                if constexpr (PASS == 0) {
                    float s = 0.f, s2 = 0.f;
#pragma unroll
                    for (int i = 0; i < 8; ++i) { s += acc[i]; s2 += acc[i] * acc[i]; }
                    s  += __shfl_xor(s, 16);
                    s2 += __shfl_xor(s2, 16);
                    if (lane < 16) { stat[w][ch] += s; stat[w][128 + ch] += s2; }
                } else {
                    const float sc = ss1[ch], sh = ss1[C1 + ch];
#pragma unroll
                    for (int i = 0; i < 8; ++i) {
                        float h = fmaxf(acc[i] * sc + sh, 0.f);
                        Hs[w][g16 * 8 + i][ch] = (__bf16)h;
                    }
                }
            }
            if constexpr (PASS >= 1) {
                // ---- layer 2: Y2[16 x C2] ----
#pragma unroll
                for (int n2 = 0; n2 < C2 / 16; ++n2) {
                    v8f acc = {0.f, 0.f, 0.f, 0.f, 0.f, 0.f, 0.f, 0.f};
#pragma unroll
                    for (int ks = 0; ks < KP2 / 32; ++ks) {
                        Frag fa, fb;
                        const __bf16* pa = &Hs[w][l16][ks * 32 + g16 * 8];
                        fa.q[0] = *(const v4u*)pa;
                        fa.q[1] = *(const v4u*)(pa + 16);
                        const __bf16* pb = &W2s[n2 * 16 + l16][ks * 32 + g16 * 8];
                        fb.q[0] = *(const v4u*)pb;
                        fb.q[1] = *(const v4u*)(pb + 16);
                        acc = __builtin_amdgcn_wmma_f32_16x16x32_bf16(
                            false, fa.v, false, fb.v, (short)0, acc, false, false);
                    }
                    const int ch = n2 * 16 + l16;
                    if constexpr (PASS == 1) {
                        float s = 0.f, s2 = 0.f;
#pragma unroll
                        for (int i = 0; i < 8; ++i) { s += acc[i]; s2 += acc[i] * acc[i]; }
                        s  += __shfl_xor(s, 16);
                        s2 += __shfl_xor(s2, 16);
                        if (lane < 16) { stat[w][ch] += s; stat[w][128 + ch] += s2; }
                    } else {
                        const float sc = ss2[ch], sh = ss2[C2 + ch];
                        float m = -1e30f;
#pragma unroll
                        for (int i = 0; i < 8; ++i)
                            m = fmaxf(m, fmaxf(acc[i] * sc + sh, 0.f));
                        m = fmaxf(m, __shfl_xor(m, 16));
                        cmax[n2] = fmaxf(cmax[n2], m);
                    }
                }
            }
        }

        if constexpr (PASS == 2) {
            if (lane < 16) {
#pragma unroll
                for (int n2 = 0; n2 < C2 / 16; ++n2) {
                    int c = COFF + n2 * 16 + l16;
                    outfeat[((size_t)b * 256 + c) * NP + p] = cmax[n2];
                }
            }
        }
    }

    if constexpr (PASS <= 1) {
        __syncthreads();
        constexpr int SC = (PASS == 0) ? C1 : C2;
        for (int c = tid; c < SC; c += 64) {
            float s  = stat[0][c]       + stat[1][c];
            float s2 = stat[0][128 + c] + stat[1][128 + c];
            statout[((size_t)blockIdx.x * SC + c) * 2 + 0] = s;
            statout[((size_t)blockIdx.x * SC + c) * 2 + 1] = s2;
        }
    }
}

// ---------------------------------------------------------------------------
// Kernel 4: deterministic reduce of block partials -> BN scale/shift
// ---------------------------------------------------------------------------
__global__ void finalize_kernel(const float* __restrict__ part, int nblocks, int cout,
                                float cnt, const float* __restrict__ g,
                                const float* __restrict__ beta, float* __restrict__ ss) {
    int c = threadIdx.x;
    if (c >= cout) return;
    float s = 0.f, s2 = 0.f;
    for (int i = 0; i < nblocks; ++i) {
        s  += part[((size_t)i * cout + c) * 2 + 0];
        s2 += part[((size_t)i * cout + c) * 2 + 1];
    }
    float mean = s / cnt;
    float var  = s2 / cnt - mean * mean;
    float sc   = g[c] * rsqrtf(var + 1e-5f);
    ss[c]        = sc;
    ss[cout + c] = beta[c] - mean * sc;
}

// ---------------------------------------------------------------------------
extern "C" void kernel_launch(void* const* d_in, const int* in_sizes, int n_in,
                              void* d_out, int out_size, void* d_ws, size_t ws_size,
                              hipStream_t stream) {
    (void)in_sizes; (void)n_in; (void)out_size; (void)ws_size;

    const float* xyz  = (const float*)d_in[0];
    const float* feat = (const float*)d_in[1];
    const float* w0_0 = (const float*)d_in[2];
    const float* g0_0 = (const float*)d_in[3];
    const float* b0_0 = (const float*)d_in[4];
    const float* w0_1 = (const float*)d_in[5];
    const float* g0_1 = (const float*)d_in[6];
    const float* b0_1 = (const float*)d_in[7];
    const float* w1_0 = (const float*)d_in[8];
    const float* g1_0 = (const float*)d_in[9];
    const float* b1_0 = (const float*)d_in[10];
    const float* w1_1 = (const float*)d_in[11];
    const float* g1_1 = (const float*)d_in[12];
    const float* b1_1 = (const float*)d_in[13];

    float* out     = (float*)d_out;
    float* newxyz  = out;                 // (B, NP, 3)
    float* outfeat = out + Bn * NP * 3;   // (B, 256, NP)

    char* ws = (char*)d_ws;
    int* fpsidx = (int*)ws;   ws += (size_t)Bn * NP * sizeof(int);
    int* idx0   = (int*)ws;   ws += (size_t)Bn * NP * 16 * sizeof(int);
    int* idx1   = (int*)ws;   ws += (size_t)Bn * NP * 32 * sizeof(int);
    float* part = (float*)ws; ws += (size_t)(Bn * NP / 8) * 128 * 2 * sizeof(float);
    float* ssA  = (float*)ws; ws += 256 * sizeof(float);
    float* ssB  = (float*)ws; ws += 256 * sizeof(float);

    // 1) FPS + new_xyz
    fps_kernel<<<Bn, 1024, 0, stream>>>(xyz, fpsidx, newxyz);

    // 2) ball queries (one wave per query point)
    ball_kernel<16><<<(Bn * NP) / 8, 256, 0, stream>>>(xyz, newxyz, idx0, 0.25f);
    ball_kernel<32><<<(Bn * NP) / 8, 256, 0, stream>>>(xyz, newxyz, idx1, 1.0f);

    constexpr int QPW  = 4;                      // query points per wave
    const int     nblk = (Bn * NP) / (2 * QPW);  // 2 waves/block

    // 3) scale 0: S=16, C1=64, C2=128, radius 0.5
    mlp_kernel<16, 64, 96, 128, 0, 0, QPW><<<nblk, 64, 0, stream>>>(
        xyz, feat, newxyz, idx0, w0_0, w0_1, nullptr, nullptr, part, nullptr);
    finalize_kernel<<<1, 64, 0, stream>>>(part, nblk, 64, (float)(Bn * NP * 16), g0_0, b0_0, ssA);
    mlp_kernel<16, 64, 96, 128, 1, 0, QPW><<<nblk, 64, 0, stream>>>(
        xyz, feat, newxyz, idx0, w0_0, w0_1, ssA, nullptr, part, nullptr);
    finalize_kernel<<<1, 128, 0, stream>>>(part, nblk, 128, (float)(Bn * NP * 16), g0_1, b0_1, ssB);
    mlp_kernel<16, 64, 96, 128, 2, 0, QPW><<<nblk, 64, 0, stream>>>(
        xyz, feat, newxyz, idx0, w0_0, w0_1, ssA, ssB, nullptr, outfeat);

    // 4) scale 1: S=32, C1=96, C2=128, radius 1.0
    mlp_kernel<32, 96, 96, 128, 0, 128, QPW><<<nblk, 64, 0, stream>>>(
        xyz, feat, newxyz, idx1, w1_0, w1_1, nullptr, nullptr, part, nullptr);
    finalize_kernel<<<1, 96, 0, stream>>>(part, nblk, 96, (float)(Bn * NP * 32), g1_0, b1_0, ssA);
    mlp_kernel<32, 96, 96, 128, 1, 128, QPW><<<nblk, 64, 0, stream>>>(
        xyz, feat, newxyz, idx1, w1_0, w1_1, ssA, nullptr, part, nullptr);
    finalize_kernel<<<1, 128, 0, stream>>>(part, nblk, 128, (float)(Bn * NP * 32), g1_1, b1_1, ssB);
    mlp_kernel<32, 96, 96, 128, 2, 128, QPW><<<nblk, 64, 0, stream>>>(
        xyz, feat, newxyz, idx1, w1_0, w1_1, ssA, ssB, nullptr, outfeat);
}
